// EHOutputPPBlock_84353157693662
// MI455X (gfx1250) — matrix operand order repaired and checked by reading, not compiled, not run on hardware
//
#include <hip/hip_runtime.h>
#include <hip/hip_bf16.h>

// ---------------------------------------------------------------------------
// Problem constants (match the reference): E=600000, N=50000, H=128, OE=256.
// E and N are taken from in_sizes at launch; H/OE are compile-time for unroll.
// ---------------------------------------------------------------------------
#define HDIM 128
#define OEDIM 256
#define NT 128          // nodes per block in the fused MLP kernel
#define LDA 264         // bf16 elements per LDS activation row (256 + 8 pad)

typedef __attribute__((ext_vector_type(16))) __bf16 v16bf;
typedef __attribute__((ext_vector_type(8)))  __bf16 v8bf;
typedef __attribute__((ext_vector_type(8)))  float  v8f;

// ---------------------------------------------------------------------------
// fp32 -> bf16 weight conversion (run once per weight matrix, tiny)
// ---------------------------------------------------------------------------
__global__ __launch_bounds__(256) void wconvert_kernel(const float* __restrict__ s,
                                                       __bf16* __restrict__ d, int n) {
    int g = blockIdx.x * 256 + threadIdx.x;
    if (g < n) d[g] = (__bf16)s[g];
}

// ---------------------------------------------------------------------------
// Edge stage: xe = (W_rbf @ rbf_e) * x_e  scatter-added into xn[i[e]].
// 32 threads per edge, 4 features per thread (float4 on x). W_rbf in LDS.
// Bandwidth-bound: ~330 MB of reads at 23.3 TB/s.
// ---------------------------------------------------------------------------
__global__ __launch_bounds__(256) void edge_scatter_kernel(
    const float* __restrict__ x, const float* __restrict__ rbf,
    const int* __restrict__ idx, const float* __restrict__ W_rbf,
    float* __restrict__ xn, int E)
{
    __shared__ float sW[HDIM * 6];
    for (int g = threadIdx.x; g < HDIM * 6; g += 256) sW[g] = W_rbf[g];
    __syncthreads();

    int g = blockIdx.x * 256 + threadIdx.x;
    int total = E * 32;
    if (g >= total) return;
    int e  = g >> 5;
    int h0 = (g & 31) * 4;

    const float* re = rbf + (size_t)e * 6;
    float r0 = re[0], r1 = re[1], r2 = re[2], r3 = re[3], r4 = re[4], r5 = re[5];
    int node = idx[e];

    float4 xv = *reinterpret_cast<const float4*>(x + (size_t)e * HDIM + h0);
    float xvv[4] = {xv.x, xv.y, xv.z, xv.w};
    float* dst = xn + (size_t)node * HDIM + h0;
#pragma unroll
    for (int j = 0; j < 4; ++j) {
        const float* wr = sW + (h0 + j) * 6;
        float w = wr[0]*r0 + wr[1]*r1 + wr[2]*r2 + wr[3]*r3 + wr[4]*r4 + wr[5]*r5;
        atomicAdd(dst + j, w * xvv[j]);   // global_atomic_add_f32, L2-resident target
    }
}

// ---------------------------------------------------------------------------
// Async global->LDS staging of one weight matrix [OEDIM x K] bf16, padded to
// row stride (K+8) bf16 so B-fragment reads rotate across all 64 LDS banks.
// Uses gfx1250 GLOBAL_LOAD_ASYNC_TO_LDS_B128 (ASYNCcnt) + S_WAIT_ASYNCCNT.
// Each lane copies independent 16B chunks; wave waits on its own ASYNCcnt,
// workgroup barrier (by the caller) makes all copies visible.
// ---------------------------------------------------------------------------
template <int K>
__device__ __forceinline__ void async_copy_weights(const __bf16* __restrict__ gw,
                                                   __bf16* lds_w, int t)
{
    constexpr int rowBytes     = K * 2;            // 256 or 512
    constexpr int chunksPerRow = rowBytes / 16;    // 16 or 32
    constexpr int ldsRowBytes  = (K + 8) * 2;      // padded LDS row stride
    constexpr int total        = OEDIM * chunksPerRow;

    unsigned lbase = (unsigned)(uintptr_t)lds_w;   // LDS byte offset (addr[31:0])
    const char* gbase = (const char*)gw;
#pragma unroll 4
    for (int c = t; c < total; c += 256) {
        int row = c / chunksPerRow;
        int ci  = c % chunksPerRow;
        unsigned    laddr = lbase + row * ldsRowBytes + ci * 16;
        const char* gaddr = gbase + row * rowBytes + ci * 16;
        asm volatile("global_load_async_to_lds_b128 %0, %1, off"
                     :: "v"(laddr), "v"(gaddr)
                     : "memory");
    }
    asm volatile("s_wait_asynccnt 0x0" ::: "memory");
}

// ---------------------------------------------------------------------------
// One dense layer on a 128-node LDS tile:  dst = act(src @ W.T + b), bf16 in/out,
// f32 WMMA accumulate. Wave w owns column tiles {2w, 2w+1}; sweeps 8 row tiles.
// A-fragment per ISA 16-bit A 16x32 layout:
//   lane<16 : row m=lane,    K = {k0..k0+7, k0+16..k0+23}
//   lane>=16: row m=lane-16, K = {k0+8..k0+15, k0+24..k0+31}
// B-fragment: col n, same K split as A (k0+8h..+7, k0+16+8h..+7) read from the
// LDS-staged weight tile (row stride K+8 -> conflict-free banking).
// ---------------------------------------------------------------------------
__device__ __forceinline__ float swishf(float v) { return v / (1.f + __expf(-v)); }

template <int K, bool ACT>
__device__ __forceinline__ void gemm_tile_layer(
    const __bf16* src, __bf16* dst, const __bf16* lds_w,
    const float* __restrict__ bias, int wave, int lane)
{
    const int hf = lane >> 4;          // lane half (K split)
    const int m  = lane & 15;
    const int col0 = wave * 32 + m;
    const int col1 = col0 + 16;
    const float bv0 = bias[col0];
    const float bv1 = bias[col1];
    const __bf16* w0 = lds_w + col0 * (K + 8);
    const __bf16* w1 = lds_w + col1 * (K + 8);

    for (int rt = 0; rt < 8; ++rt) {
        v8f acc0 = {}, acc1 = {};
        const __bf16* arow = src + (rt * 16 + m) * LDA;
#pragma unroll
        for (int k0 = 0; k0 < K; k0 += 32) {
            // A fragment from LDS (two 16B loads)
            v8bf alo = *reinterpret_cast<const v8bf*>(arow + k0 + (hf ? 8 : 0));
            v8bf ahi = *reinterpret_cast<const v8bf*>(arow + k0 + 16 + (hf ? 8 : 0));
            v16bf a;
#pragma unroll
            for (int i = 0; i < 8; ++i) { a[i] = alo[i]; a[i + 8] = ahi[i]; }
            // B fragments from LDS-staged weights (two 16B loads each)
            const int kb = k0 + (hf ? 8 : 0);
            v8bf b0lo = *reinterpret_cast<const v8bf*>(w0 + kb);
            v8bf b0hi = *reinterpret_cast<const v8bf*>(w0 + kb + 16);
            v8bf b1lo = *reinterpret_cast<const v8bf*>(w1 + kb);
            v8bf b1hi = *reinterpret_cast<const v8bf*>(w1 + kb + 16);
            v16bf b0, b1;
#pragma unroll
            for (int i = 0; i < 8; ++i) {
                b0[i] = b0lo[i]; b0[i + 8] = b0hi[i];
                b1[i] = b1lo[i]; b1[i + 8] = b1hi[i];
            }
            acc0 = __builtin_amdgcn_wmma_f32_16x16x32_bf16(false, a, false, b0,
                                                           (short)0, acc0, false, false);
            acc1 = __builtin_amdgcn_wmma_f32_16x16x32_bf16(false, a, false, b1,
                                                           (short)0, acc1, false, false);
        }
        // C/D layout: elem j -> row rt*16 + hf*8 + j, col = colTile*16 + m
#pragma unroll
        for (int j = 0; j < 8; ++j) {
            int r = rt * 16 + hf * 8 + j;
            float v0 = acc0[j] + bv0;
            float v1 = acc1[j] + bv1;
            if (ACT) { v0 = swishf(v0); v1 = swishf(v1); }
            dst[r * LDA + col0] = (__bf16)v0;
            dst[r * LDA + col1] = (__bf16)v1;
        }
    }
}

// ---------------------------------------------------------------------------
// Fused node MLP: alpha = xn@W_w.T+b_w ; y = xn@W_up.T+b_up ;
// 3x swish(y@W.T+b) ; out = (y@W_out.T)*alpha. One block = 128 nodes.
// Activations ping-pong between two bf16 LDS tiles; each layer's weights are
// async-DMA'd into a 132KB LDS tile and reused by all 8 waves x 8 row tiles.
// Total static LDS ~264.5KB of the WGP's 320KB.
// ---------------------------------------------------------------------------
__global__ __launch_bounds__(256) void fused_mlp_kernel(
    const float* __restrict__ xn,
    const __bf16* __restrict__ wup, const __bf16* __restrict__ wl1,
    const __bf16* __restrict__ wl2, const __bf16* __restrict__ wl3,
    const float* __restrict__ b_up, const float* __restrict__ b_l1,
    const float* __restrict__ b_l2, const float* __restrict__ b_l3,
    const float* __restrict__ W_out, const float* __restrict__ W_w,
    const float* __restrict__ b_w,
    float* __restrict__ out, int N)
{
    __shared__ __align__(16) __bf16 bufA[NT * LDA];
    __shared__ __align__(16) __bf16 bufB[NT * LDA];
    __shared__ __align__(16) __bf16 lds_w[OEDIM * (OEDIM + 8)];
    __shared__ float s_alpha[NT];

    const int t = threadIdx.x;
    const int lane = t & 31;
    const int wave = t >> 5;
    const int node0 = blockIdx.x * NT;

    // Stage xn tile into bufA as bf16 (coalesced f32 reads)
    for (int g = t; g < NT * HDIM; g += 256) {
        int r = g >> 7, c = g & (HDIM - 1);
        int gn = node0 + r;
        float v = (gn < N) ? xn[(size_t)gn * HDIM + c] : 0.f;
        bufA[r * LDA + c] = (__bf16)v;
    }
    // Async-stage W_up while the xn loads are in flight
    async_copy_weights<HDIM>(wup, lds_w, t);
    // alpha head (f32 precision, straight from global xn)
    if (t < NT) {
        int gn = node0 + t;
        float s = b_w[0];
        if (gn < N) {
            const float* xr = xn + (size_t)gn * HDIM;
#pragma unroll 8
            for (int k = 0; k < HDIM; ++k) s += xr[k] * W_w[k];
        }
        s_alpha[t] = s;
    }
    __syncthreads();

    gemm_tile_layer<HDIM,  false>(bufA, bufB, lds_w, b_up, wave, lane);  // lin_up
    __syncthreads();
    async_copy_weights<OEDIM>(wl1, lds_w, t);
    __syncthreads();
    gemm_tile_layer<OEDIM, true >(bufB, bufA, lds_w, b_l1, wave, lane);  // layer 1
    __syncthreads();
    async_copy_weights<OEDIM>(wl2, lds_w, t);
    __syncthreads();
    gemm_tile_layer<OEDIM, true >(bufA, bufB, lds_w, b_l2, wave, lane);  // layer 2
    __syncthreads();
    async_copy_weights<OEDIM>(wl3, lds_w, t);
    __syncthreads();
    gemm_tile_layer<OEDIM, true >(bufB, bufA, lds_w, b_l3, wave, lane);  // layer 3
    __syncthreads();

    // out = (y @ W_out.T) * alpha
    if (t < NT) {
        int gn = node0 + t;
        if (gn < N) {
            float s = 0.f;
            const __bf16* yr = bufA + t * LDA;
#pragma unroll 8
            for (int k = 0; k < OEDIM; ++k) s += (float)yr[k] * W_out[k];
            out[gn] = s * s_alpha[t];
        }
    }
}

// ---------------------------------------------------------------------------
// Host launcher
// ---------------------------------------------------------------------------
extern "C" void kernel_launch(void* const* d_in, const int* in_sizes, int n_in,
                              void* d_out, int out_size, void* d_ws, size_t ws_size,
                              hipStream_t stream)
{
    const float* x     = (const float*)d_in[0];
    const float* rbf   = (const float*)d_in[1];
    const int*   idx   = (const int*)d_in[2];
    const int*   batch = (const int*)d_in[5];
    const float* W_rbf = (const float*)d_in[7];
    const float* W_up  = (const float*)d_in[8];
    const float* b_up  = (const float*)d_in[9];
    const float* W_l1  = (const float*)d_in[10];
    const float* b_l1  = (const float*)d_in[11];
    const float* W_l2  = (const float*)d_in[12];
    const float* b_l2  = (const float*)d_in[13];
    const float* W_l3  = (const float*)d_in[14];
    const float* b_l3  = (const float*)d_in[15];
    const float* W_out = (const float*)d_in[16];
    const float* W_w   = (const float*)d_in[17];
    const float* b_w   = (const float*)d_in[18];

    const int E = in_sizes[2];   // edge count
    const int N = in_sizes[5];   // node count

    // Workspace: xn f32 [N,H], then bf16 weight copies
    float*  xn   = (float*)d_ws;
    __bf16* wupb = (__bf16*)(xn + (size_t)N * HDIM);
    __bf16* wl1b = wupb + OEDIM * HDIM;
    __bf16* wl2b = wl1b + OEDIM * OEDIM;
    __bf16* wl3b = wl2b + OEDIM * OEDIM;

    hipMemsetAsync(xn, 0, (size_t)N * HDIM * sizeof(float), stream);
    wconvert_kernel<<<(OEDIM * HDIM + 255) / 256, 256, 0, stream>>>(W_up, wupb, OEDIM * HDIM);
    wconvert_kernel<<<(OEDIM * OEDIM + 255) / 256, 256, 0, stream>>>(W_l1, wl1b, OEDIM * OEDIM);
    wconvert_kernel<<<(OEDIM * OEDIM + 255) / 256, 256, 0, stream>>>(W_l2, wl2b, OEDIM * OEDIM);
    wconvert_kernel<<<(OEDIM * OEDIM + 255) / 256, 256, 0, stream>>>(W_l3, wl3b, OEDIM * OEDIM);

    int egrid = (E * 32 + 255) / 256;
    edge_scatter_kernel<<<egrid, 256, 0, stream>>>(x, rbf, idx, W_rbf, xn, E);

    fused_mlp_kernel<<<(N + NT - 1) / NT, 256, 0, stream>>>(
        xn, wupb, wl1b, wl2b, wl3b, b_up, b_l1, b_l2, b_l3,
        W_out, W_w, b_w, (float*)d_out, N);

    // Second tuple element: batch (int32) copied bitwise after the N outputs
    hipMemcpyAsync((char*)d_out + (size_t)N * sizeof(float), batch,
                   (size_t)N * sizeof(int), hipMemcpyDeviceToDevice, stream);
}
